// F0Encoder_60026462929429
// MI455X (gfx1250) — matrix operand order repaired and checked by reading, not compiled, not run on hardware
//
#include <hip/hip_runtime.h>
#include <hip/hip_bf16.h>

typedef __bf16 bf16;
typedef __attribute__((ext_vector_type(16))) __bf16 v16bf;
typedef __attribute__((ext_vector_type(8)))  __bf16 v8bf;
typedef __attribute__((ext_vector_type(8)))  float  v8f;
typedef __attribute__((ext_vector_type(4))) unsigned int u32x4;
typedef __attribute__((ext_vector_type(4))) int i32x4;
typedef __attribute__((ext_vector_type(8))) int i32x8;

#define B_      16
#define L_      2048
#define M_      (B_ * L_)       /* 32768 rows of the activation GEMMs */
#define DE_     256
#define H_      32
#define MAXSEG_ 108             /* L/19 + 1 */
#define NSEG_   (B_ * MAXSEG_)  /* 1728 */
#define L2_     64
#define NT_     16              /* 16 column tiles of 16 -> N = 256 */

// ---------------------------------------------------------------------------
// cast f32 (m, C) -> bf16 (m, Cp) zero-padded channels
__global__ void k_cast(const float* __restrict__ src, bf16* __restrict__ dst,
                       int C, int Cp, long total) {
  long e = (long)blockIdx.x * blockDim.x + threadIdx.x;
  if (e >= total) return;
  int cp = (int)(e % Cp);
  long m = e / Cp;
  dst[e] = (cp < C) ? (bf16)src[m * C + cp] : (bf16)0.f;
}

// ---------------------------------------------------------------------------
// pack weights (O, C, KT) f32 into per-lane WMMA B-fragments:
// frag[((k*CB + cb)*NTtotal + nt0+ntl)*32 + lane][16], K-dim = C tiles of 32
__global__ void k_packw(const float* __restrict__ w, bf16* __restrict__ frag,
                        int O, int C, int Cp, int KT, int NTtotal, int nt0,
                        int NTlocal, int total) {
  int id = blockIdx.x * blockDim.x + threadIdx.x;
  if (id >= total) return;
  int lane = id & 31;
  int t = id >> 5;
  int ntl = t % NTlocal; t /= NTlocal;
  int CB = Cp / 32;
  int cb = t % CB;
  int k = t / CB;
  int col = lane & 15, half = lane >> 4;
  int o = ntl * 16 + col;
  long base = ((((long)k * CB + cb) * NTtotal + (nt0 + ntl)) * 32 + lane) * 16;
  for (int j = 0; j < 16; ++j) {
    int c = cb * 32 + half * 16 + j;            // B 32x16: lanes0-15 K=0..15, 16-31 K=16..31
    float v = (c < C && o < O) ? w[((long)o * C + c) * KT + k] : 0.f;
    frag[base + j] = (bf16)v;
  }
}

// ---------------------------------------------------------------------------
// implicit-GEMM conv / projection: Y(M, 256) = sum_k A(b, l+k-KC, :Cp) @ Wk + bias
// 8 waves/block, each wave one 16x16 tile via v_wmma_f32_16x16x32_bf16.
// The block's weight fragments (one 16-col tile, KT*CB lines of 1024B) are
// DMA'd into LDS once by the Tensor Data Mover, then reused by all waves.
template <int CB, int KT, int KC>
__global__ __launch_bounds__(256) void k_gemm(
    const bf16* __restrict__ A, const bf16* __restrict__ Wfrag,
    const float* __restrict__ bias, float* __restrict__ Y, int Cp) {
  __shared__ alignas(64) bf16 shw[KT * CB * 512];   // only LDS object -> offset 0

  const int tid = threadIdx.x;
  const int lane = tid & 31;
  const int wv = tid >> 5;
  const int mtile = blockIdx.x * 8 + wv;
  const int ntile = blockIdx.y;
  const int half = lane >> 4;
  const int m = mtile * 16 + (lane & 15);
  const int b = m >> 11;                 // / L_
  const int l = m & (L_ - 1);

#if defined(__AMDGCN__) && __has_builtin(__builtin_amdgcn_tensor_load_to_lds)
  if (wv == 0) {                         // TDM ignores EXEC; issue from one wave
    const unsigned rows = (unsigned)(KT * CB);    // tile_dim1
    const unsigned relem = 512u;                  // fragment line, bf16 elements
    unsigned long long ga =
        (unsigned long long)(uintptr_t)Wfrag + (unsigned long long)ntile * 512ull * 2ull;
    u32x4 g0;
    g0[0] = 1u;                                          // count=1 valid D#
    g0[1] = 0u;                                          // lds_addr = 0 (shw)
    g0[2] = (unsigned)(ga & 0xffffffffu);                // global_addr lo
    g0[3] = (unsigned)((ga >> 32) & 0x01ffffffu) | (2u << 30);  // addr hi | type=2
    i32x8 g1;
    g1[0] = (int)(1u << 16);                             // data_size = 2 bytes
    g1[1] = (int)((relem & 0xffffu) << 16);              // tensor_dim0[15:0]
    g1[2] = (int)(((relem >> 16) & 0xffffu) | ((rows & 0xffffu) << 16)); // dim0hi|dim1lo
    g1[3] = (int)(((rows >> 16) & 0xffffu) | (relem << 16));             // dim1hi|tile_dim0
    g1[4] = (int)(rows & 0xffffu);                       // tile_dim1 (tile_dim2=0)
    g1[5] = (int)(NT_ * 512);                            // tensor_dim0_stride lo32
    g1[6] = 0;                                           // stride hi | dim1_stride lo
    g1[7] = 0;
    i32x4 gz; gz[0] = 0; gz[1] = 0; gz[2] = 0; gz[3] = 0;
#if defined(__clang_major__) && (__clang_major__ >= 23)
    i32x8 gz8;
    gz8[0] = 0; gz8[1] = 0; gz8[2] = 0; gz8[3] = 0;
    gz8[4] = 0; gz8[5] = 0; gz8[6] = 0; gz8[7] = 0;
    __builtin_amdgcn_tensor_load_to_lds(g0, g1, gz, gz, gz8, 0);
#else
    __builtin_amdgcn_tensor_load_to_lds(g0, g1, gz, gz, 0);
#endif
    __builtin_amdgcn_s_wait_tensorcnt(0);
  }
#else
  // cooperative fallback: same LDS staging with plain loads/stores
  for (int e = tid; e < KT * CB * 512; e += 256)
    shw[e] = Wfrag[((long)(e >> 9) * NT_ + ntile) * 512 + (e & 511)];
#endif
  __syncthreads();

  v8f acc;
#pragma unroll
  for (int v = 0; v < 8; ++v) acc[v] = 0.f;

  for (int k = 0; k < KT; ++k) {
    const int l2 = l + k - KC;
    const bool ok = (l2 >= 0) && (l2 < L_);
    const bf16* arow = A + (long)(b * L_ + (ok ? l2 : 0)) * Cp;
    if (ok) __builtin_prefetch(arow, 0, 0);   // global_prefetch_b8
#pragma unroll
    for (int cb = 0; cb < CB; ++cb) {
      v16bf a;
      if (ok) {
        // A order used here (same as packed B): lane half h holds K = h*16..h*16+15
        const bf16* p = arow + cb * 32 + half * 16;
        v8bf lo = *(const v8bf*)(p);
        v8bf hi = *(const v8bf*)(p + 8);
#pragma unroll
        for (int j = 0; j < 8; ++j) { a[j] = lo[j]; a[8 + j] = hi[j]; }
      } else {
#pragma unroll
        for (int j = 0; j < 16; ++j) a[j] = (bf16)0.f;
      }
      v16bf bw = *(const v16bf*)(shw + ((k * CB + cb) * 32 + lane) * 16);
      acc = __builtin_amdgcn_wmma_f32_16x16x32_bf16(
          false, a, false, bw, (short)0, acc, false, false);
    }
  }
  const int ncol = ntile * 16 + (lane & 15);
  const float bv = bias ? bias[ncol] : 0.f;
#pragma unroll
  for (int v = 0; v < 8; ++v) {                 // C/D: VGPR v -> M=v (+8 upper half)
    int mo = mtile * 16 + v + half * 8;
    Y[(long)mo * 256 + ncol] = acc[v] + bv;
  }
}

// ---------------------------------------------------------------------------
// training-mode BN statistics: one block per channel, LDS tree reduction
__global__ __launch_bounds__(256) void k_bnstats(const float* __restrict__ Y,
                                                 float* __restrict__ stats) {
  const int c = blockIdx.x, tid = threadIdx.x;
  float s = 0.f, s2 = 0.f;
  for (int m = tid; m < M_; m += 256) {
    float v = Y[(long)m * DE_ + c];
    s += v; s2 += v * v;
  }
  __shared__ float rs[256], rq[256];
  rs[tid] = s; rq[tid] = s2;
  __syncthreads();
  for (int st = 128; st > 0; st >>= 1) {
    if (tid < st) { rs[tid] += rs[tid + st]; rq[tid] += rq[tid + st]; }
    __syncthreads();
  }
  if (tid == 0) {
    float mean = rs[0] / (float)M_;
    float var = rq[0] / (float)M_ - mean * mean;   // biased var
    stats[c] = mean;
    stats[DE_ + c] = rsqrtf(var + 1e-5f);
  }
}

__global__ void k_bnrelu(const float* __restrict__ Y, const float* __restrict__ stats,
                         const float* __restrict__ g, const float* __restrict__ be,
                         float* __restrict__ Hout) {
  long e = (long)blockIdx.x * 256 + threadIdx.x;
  if (e >= (long)M_ * DE_) return;
  int c = (int)(e & (DE_ - 1));
  float v = (Y[e] - stats[c]) * stats[DE_ + c] * g[c] + be[c];
  Hout[e] = v > 0.f ? v : 0.f;
}

// ---------------------------------------------------------------------------
// interp_lnr metadata: per-segment valid count + segment start offset in source
__global__ void k_segmeta(const float* __restrict__ scales, const int* __restrict__ lens,
                          int layer, int* __restrict__ counts, int* __restrict__ offsets) {
  int n = blockIdx.x * blockDim.x + threadIdx.x;
  if (n >= NSEG_) return;
  int b = n / MAXSEG_, s = n % MAXSEG_;
  const int* lb = lens + layer * NSEG_ + b * MAXSEG_;
  int off = 0;
  for (int t = 0; t < s; ++t) off += lb[t];
  float scale = scales[layer * NSEG_ + n] + 0.5f;
  float limseg = (float)(lb[s] - 1);
  float limorg = (float)(L_ - 1 - off);
  int cnt = 0;
  for (int j = 0; j < L2_; ++j) {
    float fl = floorf((float)j / scale);
    if (fl < limseg && fl < limorg) ++cnt;        // mask is prefix-true in j
  }
  counts[n] = cnt;
  offsets[n] = off;
}

// per-batch exclusive scan of segment counts (108 segments, trivial)
__global__ void k_segbase(const int* __restrict__ counts, int* __restrict__ bases) {
  int b = threadIdx.x;
  if (b >= B_) return;
  int acc = 0;
  for (int s = 0; s < MAXSEG_; ++s) {
    bases[b * MAXSEG_ + s] = acc;
    acc += counts[b * MAXSEG_ + s];
  }
}

__global__ void k_zero(float* __restrict__ p, long n) {
  for (long i = (long)blockIdx.x * blockDim.x + threadIdx.x; i < n;
       i += (long)gridDim.x * blockDim.x)
    p[i] = 0.f;
}

// compacting linear-interp resample: one block per segment, one thread per channel
__global__ __launch_bounds__(256) void k_interp(
    const float* __restrict__ H, const float* __restrict__ scales, int layer,
    const int* __restrict__ counts, const int* __restrict__ offsets,
    const int* __restrict__ bases, float* __restrict__ T) {
  const int n = blockIdx.x;
  const int c = threadIdx.x;
  const int b = n / MAXSEG_;
  const float scale = scales[layer * NSEG_ + n] + 0.5f;
  const int cnt = counts[n];
  const int off = offsets[n];
  const int base = bases[n];
  const long rb = (long)b * L_ * DE_;
  for (int j = 0; j < cnt; ++j) {
    const int pos = base + j;
    if (pos >= L_) break;                          // overflow rows discarded
    const float is = (float)j / scale;
    const float fl = floorf(is);
    const float lam = is - fl;
    int io = (int)fl + off;
    int ifl = min(max(io, 0), L_ - 1);
    int icl = min(ifl + 1, L_ - 1);
    float yfl = H[rb + (long)ifl * DE_ + c];
    float ycl = H[rb + (long)icl * DE_ + c];
    T[rb + (long)pos * DE_ + c] = (1.f - lam) * yfl + lam * ycl;
  }
}

// ---------------------------------------------------------------------------
// LSTM: combined fwd/bwd gate bias
__global__ void k_lstmbias(const float* __restrict__ bf1, const float* __restrict__ bf2,
                           const float* __restrict__ bb1, const float* __restrict__ bb2,
                           float* __restrict__ bias) {
  int n = threadIdx.x;
  bias[n] = (n < 128) ? (bf1[n] + bf2[n]) : (bb1[n - 128] + bb2[n - 128]);
}

// pack Whh (128, 32) into B-fragments for the K=32 recurrent WMMA (one dir)
__global__ void k_packhh(const float* __restrict__ whh, bf16* __restrict__ frag) {
  int tid = threadIdx.x;                 // 256 threads = 8 ntiles * 32 lanes
  int lane = tid & 31, nt = tid >> 5;
  int col = lane & 15, half = lane >> 4;
  int n = nt * 16 + col;
#pragma unroll
  for (int j = 0; j < 16; ++j) {
    int k = half * 16 + j;
    frag[((long)(nt * 32 + lane)) * 16 + j] = (bf16)whh[n * 32 + k];
  }
}

// persistent bidirectional LSTM: block 0 = forward, block 1 = backward.
// 8 waves, each owns a 16(batch)x16(gate) tile; h/c live in LDS; one
// v_wmma_f32_16x16x32_bf16 (K = H = 32) per wave per timestep.
__global__ __launch_bounds__(256) void k_lstm(const float* __restrict__ xg,
                                              const bf16* __restrict__ fragHH,
                                              float* __restrict__ out) {
  const int dir = blockIdx.x;
  const int tid = threadIdx.x;
  const int lane = tid & 31;
  const int wv = tid >> 5;
  const int col = lane & 15;
  const int half = lane >> 4;
  const int gcol = wv * 16 + col;        // gate column within direction (0..127)

  __shared__ alignas(64) bf16 sh_h[16 * 32];
  __shared__ float sh_c[16 * 32];
  __shared__ float sh_g[16 * 128];

  for (int e = tid; e < 512; e += 256) { sh_c[e] = 0.f; sh_h[e] = (bf16)0.f; }
  __syncthreads();

  v16bf bw = *(const v16bf*)(fragHH + ((long)(dir * 8 + wv) * 32 + lane) * 16);

  for (int s = 0; s < L_; ++s) {
    const int tt = (dir == 0) ? s : (L_ - 1 - s);
    v8f acc;
#pragma unroll
    for (int v = 0; v < 8; ++v) {
      int br = v + half * 8;             // batch row for this accumulator lane
      acc[v] = xg[((long)br * L_ + tt) * 256 + dir * 128 + gcol];
    }
    v16bf a;                              // A = h (16x32 bf16) from LDS
    {
      const bf16* hp = sh_h + (lane & 15) * 32 + half * 16;
#pragma unroll
      for (int j = 0; j < 8; ++j) { a[j] = hp[j]; a[8 + j] = hp[8 + j]; }
    }
    acc = __builtin_amdgcn_wmma_f32_16x16x32_bf16(
        false, a, false, bw, (short)0, acc, false, false);
#pragma unroll
    for (int v = 0; v < 8; ++v) {
      int br = v + half * 8;
      sh_g[br * 128 + gcol] = acc[v];
    }
    __syncthreads();
    for (int e = tid; e < 512; e += 256) {  // 16 batch x 32 hidden
      int r = e >> 5, hc = e & 31;
      float gi = sh_g[r * 128 + hc];
      float gf = sh_g[r * 128 + 32 + hc];
      float gg = sh_g[r * 128 + 64 + hc];
      float go = sh_g[r * 128 + 96 + hc];
      float si = 1.f / (1.f + __expf(-gi));
      float sf = 1.f / (1.f + __expf(-gf));
      float tg = tanhf(gg);
      float so = 1.f / (1.f + __expf(-go));
      float cc = sf * sh_c[e] + si * tg;
      sh_c[e] = cc;
      float hh = so * tanhf(cc);
      sh_h[r * 32 + hc] = (bf16)hh;
      if (dir == 0) {
        if ((tt & 7) == 7) out[((long)r * 256 + (tt >> 3)) * 64 + hc] = hh;
      } else {
        if ((tt & 7) == 0) out[((long)r * 256 + (tt >> 3)) * 64 + 32 + hc] = hh;
      }
    }
    __syncthreads();
  }
}

// ---------------------------------------------------------------------------
extern "C" void kernel_launch(void* const* d_in, const int* in_sizes, int n_in,
                              void* d_out, int out_size, void* d_ws, size_t ws_size,
                              hipStream_t stream) {
  (void)in_sizes; (void)n_in; (void)out_size; (void)ws_size;

  const float* x      = (const float*)d_in[0];
  const float* scales = (const float*)d_in[1];
  const int*   lens   = (const int*)d_in[2];
  const float* W[3]   = {(const float*)d_in[3],  (const float*)d_in[7],  (const float*)d_in[11]};
  const float* Bc[3]  = {(const float*)d_in[4],  (const float*)d_in[8],  (const float*)d_in[12]};
  const float* G[3]   = {(const float*)d_in[5],  (const float*)d_in[9],  (const float*)d_in[13]};
  const float* Be[3]  = {(const float*)d_in[6],  (const float*)d_in[10], (const float*)d_in[14]};
  const float* wih_f  = (const float*)d_in[15];
  const float* whh_f  = (const float*)d_in[16];
  const float* bih_f  = (const float*)d_in[17];
  const float* bhh_f  = (const float*)d_in[18];
  const float* wih_b  = (const float*)d_in[19];
  const float* whh_b  = (const float*)d_in[20];
  const float* bih_b  = (const float*)d_in[21];
  const float* bhh_b  = (const float*)d_in[22];
  float* out = (float*)d_out;

  char* base = (char*)d_ws;
  size_t off = 0;
  auto take = [&](size_t bytes) -> void* {
    off = (off + 255) & ~(size_t)255;
    void* p = base + off;
    off += bytes;
    return p;
  };
  bf16*  actbf   = (bf16*)take((size_t)M_ * 288 * sizeof(bf16));
  float* convY   = (float*)take((size_t)M_ * DE_ * sizeof(float));  // reused as xg
  float* bnH     = (float*)take((size_t)M_ * DE_ * sizeof(float));
  float* interpT = (float*)take((size_t)M_ * DE_ * sizeof(float));
  float* stats   = (float*)take(2 * DE_ * sizeof(float));
  int*   counts  = (int*)take(NSEG_ * sizeof(int));
  int*   offsets = (int*)take(NSEG_ * sizeof(int));
  int*   basesv  = (int*)take(NSEG_ * sizeof(int));
  bf16*  wfrag   = (bf16*)take((size_t)5 * 9 * 16 * 32 * 16 * sizeof(bf16));
  bf16*  wfragIH = (bf16*)take((size_t)8 * 16 * 32 * 16 * sizeof(bf16));
  bf16*  fragHH  = (bf16*)take((size_t)2 * 8 * 32 * 16 * sizeof(bf16));
  float* biasL   = (float*)take(256 * sizeof(float));

  for (int i = 0; i < 3; ++i) {
    const int C = (i == 0) ? 257 : 256;
    const int Cp = (i == 0) ? 288 : 256;
    const int CB = Cp / 32;
    const float* src = (i == 0) ? x : interpT;

    long ctot = (long)M_ * Cp;
    k_cast<<<(int)((ctot + 255) / 256), 256, 0, stream>>>(src, actbf, C, Cp, ctot);

    int ptot = 5 * CB * 16 * 32;
    k_packw<<<(ptot + 255) / 256, 256, 0, stream>>>(W[i], wfrag, 256, C, Cp, 5, 16, 0, 16, ptot);

    if (i == 0)
      k_gemm<9, 5, 2><<<dim3(M_ / 128, NT_), 256, 0, stream>>>(actbf, wfrag, Bc[i], convY, Cp);
    else
      k_gemm<8, 5, 2><<<dim3(M_ / 128, NT_), 256, 0, stream>>>(actbf, wfrag, Bc[i], convY, Cp);

    k_bnstats<<<DE_, 256, 0, stream>>>(convY, stats);
    k_bnrelu<<<(int)(((long)M_ * DE_ + 255) / 256), 256, 0, stream>>>(convY, stats, G[i], Be[i], bnH);

    k_segmeta<<<(NSEG_ + 255) / 256, 256, 0, stream>>>(scales, lens, i, counts, offsets);
    k_segbase<<<1, 32, 0, stream>>>(counts, basesv);
    k_zero<<<4096, 256, 0, stream>>>(interpT, (long)M_ * DE_);
    k_interp<<<NSEG_, 256, 0, stream>>>(bnH, scales, i, counts, offsets, basesv, interpT);
  }

  // ---- LSTM input projection: xg = seqs @ [Wih_f ; Wih_b]^T + (bih+bhh) ----
  long ctot = (long)M_ * 256;
  k_cast<<<(int)((ctot + 255) / 256), 256, 0, stream>>>(interpT, actbf, 256, 256, ctot);

  int ptot = 1 * 8 * 8 * 32;
  k_packw<<<(ptot + 255) / 256, 256, 0, stream>>>(wih_f, wfragIH, 128, 256, 256, 1, 16, 0, 8, ptot);
  k_packw<<<(ptot + 255) / 256, 256, 0, stream>>>(wih_b, wfragIH, 128, 256, 256, 1, 16, 8, 8, ptot);
  k_lstmbias<<<1, 256, 0, stream>>>(bih_f, bhh_f, bih_b, bhh_b, biasL);

  k_gemm<8, 1, 0><<<dim3(M_ / 128, NT_), 256, 0, stream>>>(actbf, wfragIH, biasL, convY, 256);

  k_packhh<<<1, 256, 0, stream>>>(whh_f, fragHH);
  k_packhh<<<1, 256, 0, stream>>>(whh_b, fragHH + 8 * 32 * 16);

  k_lstm<<<2, 256, 0, stream>>>(convY, fragHH, out);
}
